// test_LSTM_33947421507695
// MI455X (gfx1250) — matrix operand order repaired and checked by reading, not compiled, not run on hardware
//
#include <hip/hip_runtime.h>
#include <hip/hip_bf16.h>

typedef __attribute__((ext_vector_type(16))) __bf16 v16bf;
typedef __attribute__((ext_vector_type(8)))  float  v8f;

#define B_   64
#define S_   512
#define E_   512
#define H_   1024
#define GN   8192      /* K * 4H */
#define KD   1536      /* E + H  */
#define KCHUNKS 48     /* KD / 32 */
#define LDSTRIDE 1544  /* KD + 8 pad (breaks LDS bank conflicts) */
#define GSTRIDE  132   /* gate-exchange stride in floats (132 % 64 banks = 4) */

// ---------------------------------------------------------------------------
// One-time prep: swizzle Wcat = [W_ih | W_hh] (fp32) into bf16 WMMA B-fragment
// layout. Element order: (((ntile*48 + kchunk)*32 + lane)*16 + j), where
// lane holds B[k][n] with n = ntile*16 + lane[3:0], k = kchunk*32 + lane[4]*16 + j.
// Each lane's 16 bf16 (32B) are contiguous -> two coalesced b128 loads at use.
// ---------------------------------------------------------------------------
__global__ void prep_wfrag(const float* __restrict__ Wih,
                           const float* __restrict__ Whh,
                           __bf16* __restrict__ Wfrag) {
  size_t idx = (size_t)blockIdx.x * 256 + threadIdx.x;
  if (idx >= (size_t)GN * KD) return;
  int j    = idx & 15;
  int lane = (idx >> 4) & 31;
  size_t t = idx >> 9;
  int kc = (int)(t % KCHUNKS);
  int nt = (int)(t / KCHUNKS);
  int n = nt * 16 + (lane & 15);
  int k = kc * 32 + (lane >> 4) * 16 + j;
  int cell = n >> 12;        // 0..1
  int g    = n & 4095;       // 0..4H-1
  float v = (k < E_) ? Wih[((size_t)cell * 4096 + g) * E_ + k]
                     : Whh[((size_t)cell * 4096 + g) * H_ + (k - E_)];
  Wfrag[idx] = (__bf16)v;
}

__global__ void prep_xbf(const float* __restrict__ x, __bf16* __restrict__ xbf,
                         size_t n) {
  size_t i = (size_t)blockIdx.x * 256 + threadIdx.x;
  if (i < n) xbf[i] = (__bf16)x[i];
}

__global__ void prep_state(const float* __restrict__ bih,
                           const float* __restrict__ bhh,
                           float* __restrict__ biasc,
                           __bf16* __restrict__ Hbf,
                           float* __restrict__ Cst) {
  int i = blockIdx.x * 256 + threadIdx.x;   // B_*H_ = 65536 threads
  if (i < GN) biasc[i] = bih[i] + bhh[i];
  Hbf[i] = (__bf16)0.0f;
  Cst[i] = 0.0f;
}

// ---------------------------------------------------------------------------
// Fused per-step kernel: bf16 WMMA GEMM for gates + LSTM nonlinearity.
//
// Block bx (64 blocks): cell = bx>>5, jbase = (bx&31)*32.
// The block's 128 N-columns are { cell*4096 + gate*1024 + jbase + jj },
// gate in 0..3, jj in 0..31 -- i.e. ALL FOUR GATES of one cell for a 32-wide
// j range, so the nonlinearity can be applied in-block after an LDS exchange.
// Wave w (8 waves): gate = w>>1, jsub = (w&1)*16 -> one contiguous 16-col
// N-tile; all four 16-row M tiles (batch = 64).
//
// Phase 1: stage whole activation [64 x 1536] bf16 in LDS (192KB < 320KB/WGP),
//          barrier-free K loop of 48 x (2 global b128 + 8 ds b128 + 4 WMMA).
// Phase 2: dump accumulators to LDS (aliasing the A tile), barrier.
// Phase 3: threads with cid[b]==cell apply gates, update Cst/Hbf, write outs.
// ---------------------------------------------------------------------------
__global__ __launch_bounds__(256)
void lstm_step(const __bf16* __restrict__ Xbf,   // [S,B,E]
               const __bf16* __restrict__ Hbf,   // [B,H] (prev h, bf16)
               const __bf16* __restrict__ Wfrag, // swizzled B-fragments
               const float* __restrict__ biasc,  // [GN]
               const int*   __restrict__ ids,    // [B,S]
               __bf16* __restrict__ Hout,        // [B,H] (next h, bf16)
               float*  __restrict__ Cst,         // [B,H] fp32 cell state
               float*  __restrict__ out,         // outs | hT | cT
               int s) {
  extern __shared__ char smem[];
  __bf16* As = (__bf16*)smem;                    // [64][LDSTRIDE] bf16
  float*  Gs = (float*)smem;                     // [64][GSTRIDE] fp32 (aliased)
  const int tid = threadIdx.x;

  // ---- Phase 1a: cooperative A fill (16B chunks, 48 per thread) ----
  const __bf16* xrow = Xbf + (size_t)s * B_ * E_;
  for (int i = 0; i < 48; ++i) {
    int q  = tid + i * 256;
    int m  = q / 192;                // 192 chunks per row (1536/8)
    int kk = (q % 192) * 8;
    const __bf16* src = (kk < E_) ? (xrow + m * E_ + kk)
                                  : (Hbf + m * H_ + (kk - E_));
    *(uint4*)(&As[m * LDSTRIDE + kk]) = *(const uint4*)src;
  }
  __syncthreads();

  const int lane  = tid & 31;
  const int wave  = tid >> 5;
  const int cell  = blockIdx.x >> 5;
  const int jbase = (blockIdx.x & 31) * 32;
  const int gate  = wave >> 1;
  const int jsub  = (wave & 1) * 16;
  const int col16 = cell * 4096 + gate * 1024 + jbase + jsub; // 16-aligned
  const int ntile = col16 >> 4;
  const int mrow  = lane & 15;            // A: M = lane[3:0]
  const int kbase = (lane >> 4) * 16;     // A: K half by lane[4]

  v8f acc[4] = {};

  // B-fragment stream: 512 bf16 (=64 uint4) per (ntile,kchunk).
  const uint4* wp =
      (const uint4*)(Wfrag + (size_t)ntile * KCHUNKS * 512) + lane * 2;

  for (int kc = 0; kc < KCHUNKS; ++kc) {
    union { uint4 u[2]; v16bf v; } b;
    b.u[0] = wp[0];
    b.u[1] = wp[1];
    wp += 64;
    if (kc + 1 < KCHUNKS) __builtin_prefetch(wp, 0, 0);

#pragma unroll
    for (int mt = 0; mt < 4; ++mt) {
      union { uint4 u[2]; v16bf v; } a;
      const __bf16* ap = &As[(mt * 16 + mrow) * LDSTRIDE + kc * 32 + kbase];
      a.u[0] = *(const uint4*)ap;          // K = kbase .. kbase+7
      a.u[1] = *(const uint4*)(ap + 8);    // K = kbase+8 .. kbase+15
      acc[mt] = __builtin_amdgcn_wmma_f32_16x16x32_bf16(
          false, a.v, false, b.v, (short)0, acc[mt], false, false);
    }
  }

  // ---- Phase 2: exchange accumulators through LDS (alias A tile) ----
  __syncthreads();   // all waves done reading As before overwriting with Gs
  // C/D layout: VGPR r -> M = r + 8*lane[4], N = lane[3:0]
  const int lcol  = wave * 16 + (lane & 15);   // = gate*32 + jj
  const int rbase = (lane >> 4) * 8;
#pragma unroll
  for (int mt = 0; mt < 4; ++mt)
#pragma unroll
    for (int r = 0; r < 8; ++r)
      Gs[(mt * 16 + rbase + r) * GSTRIDE + lcol] = acc[mt][r];
  __syncthreads();

  // ---- Phase 3: LSTM update for (b, j) owned by this block ----
  // 64 batches x 32 j's = 2048 elements, 8 per thread.
#pragma unroll
  for (int t = 0; t < 8; ++t) {
    int idx = tid + t * 256;
    int b  = idx >> 5;
    int jj = idx & 31;
    int cid = ids[b * S_ + s] & 1;       // token % 2 (ids >= 0)
    if (cid != cell) continue;           // the other cell's block handles it
    int j = jbase + jj;
    const float* gsrow = Gs + b * GSTRIDE;
    const float* bb = biasc + cell * 4096 + j;
    float gi = gsrow[0 * 32 + jj] + bb[0];
    float gf = gsrow[1 * 32 + jj] + bb[1024];
    float gg = gsrow[2 * 32 + jj] + bb[2048];
    float go = gsrow[3 * 32 + jj] + bb[3072];
    float si = 1.0f / (1.0f + __expf(-gi));
    float sf = 1.0f / (1.0f + __expf(-gf));
    float so = 1.0f / (1.0f + __expf(-go));
    int hidx = b * H_ + j;
    float c = sf * Cst[hidx] + si * tanhf(gg);
    float h = so * tanhf(c);
    Cst[hidx] = c;
    Hout[hidx] = (__bf16)h;
    out[(size_t)s * B_ * H_ + hidx] = h;
    if (s == S_ - 1) {
      out[(size_t)S_ * B_ * H_ + hidx] = h;               // hT
      out[(size_t)S_ * B_ * H_ + B_ * H_ + hidx] = c;     // cT
    }
  }
}

extern "C" void kernel_launch(void* const* d_in, const int* in_sizes, int n_in,
                              void* d_out, int out_size, void* d_ws, size_t ws_size,
                              hipStream_t stream) {
  const int*   ids = (const int*)d_in[0];
  const float* x   = (const float*)d_in[1];
  const float* Wih = (const float*)d_in[2];
  const float* Whh = (const float*)d_in[3];
  const float* bih = (const float*)d_in[4];
  const float* bhh = (const float*)d_in[5];
  float* out = (float*)d_out;

  char* ws = (char*)d_ws;
  __bf16* Wfrag = (__bf16*)ws;  ws += (size_t)GN * KD * sizeof(__bf16);      // 25.2 MB
  __bf16* Xbf   = (__bf16*)ws;  ws += (size_t)S_ * B_ * E_ * sizeof(__bf16); // 33.6 MB
  float*  biasc = (float*)ws;   ws += (size_t)GN * sizeof(float);
  __bf16* Hbf   = (__bf16*)ws;  ws += (size_t)B_ * H_ * sizeof(__bf16);
  float*  Cst   = (float*)ws;   // 256 KB

  size_t nw = (size_t)GN * KD;
  prep_wfrag<<<(unsigned)((nw + 255) / 256), 256, 0, stream>>>(Wih, Whh, Wfrag);
  size_t nx = (size_t)S_ * B_ * E_;
  prep_xbf<<<(unsigned)((nx + 255) / 256), 256, 0, stream>>>(x, Xbf, nx);
  prep_state<<<(B_ * H_) / 256, 256, 0, stream>>>(bih, bhh, biasc, Hbf, Cst);

  size_t shmem = (size_t)64 * LDSTRIDE * sizeof(__bf16);  // 192.6 KB dynamic LDS
  for (int s = 0; s < S_; ++s) {
    lstm_step<<<64, 256, shmem, stream>>>(Xbf, Hbf, Wfrag, biasc, ids, Hbf,
                                          Cst, out, s);
  }
}
